// Attention_4612794875918
// MI455X (gfx1250) — matrix operand order repaired and checked by reading, not compiled, not run on hardware
//
#include <hip/hip_runtime.h>

// Problem constants (from reference)
#define B_  4
#define NH_ 8
#define T_  2048
#define N_  256
#define D_  512
#define BT_ (B_ * T_)

typedef __attribute__((ext_vector_type(8)))  __bf16 v8bf;
typedef __attribute__((ext_vector_type(16))) __bf16 v16bf;
typedef __attribute__((ext_vector_type(8)))  float  v8f;

#if defined(__has_builtin)
#if __has_builtin(__builtin_amdgcn_global_load_async_to_lds_b32)
#define HAVE_ASYNC_LDS 1
#endif
#endif

// ---------------------------------------------------------------------------
// helpers
// ---------------------------------------------------------------------------
__device__ __forceinline__ unsigned bf16pair(float a, float b) {
  __bf16 x = (__bf16)a, y = (__bf16)b;
  unsigned short ux = __builtin_bit_cast(unsigned short, x);
  unsigned short uy = __builtin_bit_cast(unsigned short, y);
  return (unsigned)ux | ((unsigned)uy << 16);
}

__device__ __forceinline__ v16bf cat16(v8bf lo, v8bf hi) {
  return __builtin_shufflevector(lo, hi, 0,1,2,3,4,5,6,7,8,9,10,11,12,13,14,15);
}

// A fragment (16x32 bf16, row-major row in LDS, interleaved K layout per ISA:
// lane half h, element e<8 -> K = h*8+e ; e>=8 -> K = 16+h*8+(e-8))
__device__ __forceinline__ v16bf load_fragA(const __bf16* rowbase, int half) {
  v8bf lo = *(const v8bf*)(rowbase + half * 8);
  v8bf hi = *(const v8bf*)(rowbase + 16 + half * 8);
  return cat16(lo, hi);
}
// B fragment (32x16 bf16; same interleaved K layout as A: e<8 -> K = h*8+e ; e>=8 -> K = 16+h*8+(e-8))
__device__ __forceinline__ v16bf load_fragB(const __bf16* rowbase, int half) {
  v8bf lo = *(const v8bf*)(rowbase + half * 8);
  v8bf hi = *(const v8bf*)(rowbase + 16 + half * 8);
  return cat16(lo, hi);
}

__device__ __forceinline__ v8f wmma_bf16(v16bf a, v16bf b, v8f c) {
  return __builtin_amdgcn_wmma_f32_16x16x32_bf16(false, a, false, b, (short)0, c,
                                                 false, false);
}

// f32 -> bf16 tile, row-major (R rows x C cols), C % 4 == 0
template <int R, int C, int NT>
__device__ __forceinline__ void load_tile(__bf16* lds, const float* src,
                                          int stride, int tid) {
  constexpr int TOT = R * (C / 4);
  for (int e = tid; e < TOT; e += NT) {
    int row = e / (C / 4);
    int c4  = (e % (C / 4)) * 4;
    float4 v = *(const float4*)(src + (size_t)row * stride + c4);
    uint2 w;
    w.x = bf16pair(v.x, v.y);
    w.y = bf16pair(v.z, v.w);
    *(uint2*)(lds + (size_t)row * C + c4) = w;
  }
}

// f32 (R rows x C cols) -> bf16 transposed into LDS as [C][R]
template <int R, int C, int NT>
__device__ __forceinline__ void load_tile_T(__bf16* lds, const float* src,
                                            int stride, int tid) {
  constexpr int TOT = R * (C / 4);
  for (int e = tid; e < TOT; e += NT) {
    int row = e / (C / 4);
    int c4  = (e % (C / 4)) * 4;
    float4 v = *(const float4*)(src + (size_t)row * stride + c4);
    lds[(c4 + 0) * R + row] = (__bf16)v.x;
    lds[(c4 + 1) * R + row] = (__bf16)v.y;
    lds[(c4 + 2) * R + row] = (__bf16)v.z;
    lds[(c4 + 3) * R + row] = (__bf16)v.w;
  }
}

// RoPE'd (NT/8)x256 tile loader: row = tid/8, 32 features per thread.
// freq[n] = THETA^(-2*floor(n/2)/N)/(2pi) = 2^(-q/16)/(2pi), q = even feature.
// Phase fraction fr is in turns; hardware v_sin/v_cos evaluate sin/cos(2*pi*x),
// so fr feeds __builtin_amdgcn_{sinf,cosf} directly (no libm range reduction).
__device__ __forceinline__ void load_rope_tile(__bf16* lds, const float* src,
                                               int t0, int tid) {
  int row = tid >> 3;
  int f0  = (tid & 7) * 32;
  float t = (float)(t0 + row);
  const float* p = src + (size_t)row * N_ + f0;
#pragma unroll
  for (int g = 0; g < 4; ++g) {
    unsigned cc[4];
#pragma unroll
    for (int j = 0; j < 4; ++j) {
      int pr = g * 4 + j;
      float2 xy = *(const float2*)(p + 2 * pr);
      int f = f0 + 2 * pr;
      float freq = exp2f(-(float)f * 0.0625f) * 0.15915494309189535f;
      float ph = t * freq;
      float fr = ph - floorf(ph);             // phase in turns, [0,1)
      float cs = __builtin_amdgcn_cosf(fr);   // cos(2*pi*fr)
      float sn = __builtin_amdgcn_sinf(fr);   // sin(2*pi*fr)
      cc[j] = bf16pair(xy.x * cs - xy.y * sn, xy.y * cs + xy.x * sn);
    }
    uint4 w;
    w.x = cc[0]; w.y = cc[1]; w.z = cc[2]; w.w = cc[3];
    *(uint4*)(lds + (size_t)row * N_ + f0 + g * 8) = w;
  }
}

// ---------------------------------------------------------------------------
// Kernel A: fused causal QR.QR^T @ V  ->  out = (1-gate) * y_standard
// grid (T/64, B*NH), 512 threads (16 waves). 64x64 tiles, 128KB LDS.
// Wave w: output tile (mtile = w&3, dquad = w>>2); S tile (smt = w&3, jt = w>>2).
// ---------------------------------------------------------------------------
__global__ __launch_bounds__(512) void attn_kernel(const float* __restrict__ Q,
                                                   const float* __restrict__ V,
                                                   const float* __restrict__ mgate,
                                                   float* __restrict__ out) {
  __shared__ __align__(16) __bf16 lds_q[64 * 256];   // 32KB  QR row tile
  __shared__ __align__(16) __bf16 lds_k[64 * 256];   // 32KB  QR col tile (reused for S)
  __shared__ __align__(16) __bf16 lds_vt[512 * 64];  // 64KB  V^T [d][s]

  int tid = threadIdx.x, lane = tid & 31, wave = tid >> 5;
  int m = lane & 15, half = lane >> 4;
  int bh = blockIdx.y;
  int i0 = blockIdx.x * 64;
  const float* Qbh = Q + (size_t)bh * T_ * N_;
  const float* Vbh = V + (size_t)bh * T_ * D_;

  int mtile = wave & 3;
  int dquad = wave >> 2;

  v8f acc[8];
#pragma unroll
  for (int i = 0; i < 8; ++i) acc[i] = (v8f){};

  load_rope_tile(lds_q, Qbh + (size_t)i0 * N_, i0, tid);

  for (int j0 = 0; j0 <= i0; j0 += 64) {
    __syncthreads();  // prior iter's LDS reads done before overwrite
    load_rope_tile(lds_k, Qbh + (size_t)j0 * N_, j0, tid);
    load_tile_T<64, 512, 512>(lds_vt, Vbh + (size_t)j0 * D_, D_, tid);
    // prefetch next K/V tile while this one is consumed
    if (j0 + 64 <= i0) {
      __builtin_prefetch(Qbh + (size_t)(j0 + 64) * N_ + tid * 32, 0, 1);
      __builtin_prefetch(Vbh + (size_t)(j0 + 64) * D_ + tid * 64, 0, 1);
    }
    __syncthreads();

    // ---- S = Q_tile . K_tile^T  (16 waves own the sixteen 16x16 S tiles) ----
    v8f sacc = (v8f){};
    int smt = wave & 3, jt = wave >> 2;
#pragma unroll
    for (int k0 = 0; k0 < N_; k0 += 32) {
      v16bf a  = load_fragA(lds_q + (smt * 16 + m) * N_ + k0, half);
      v16bf bm = load_fragB(lds_k + (jt * 16 + m) * N_ + k0, half);
      sacc = wmma_bf16(a, bm, sacc);
    }
    __syncthreads();  // all K reads complete before S overwrites lds_k
    if (j0 == i0) {   // strict-lower causal mask only hits the diagonal tile
#pragma unroll
      for (int r = 0; r < 8; ++r) {
        int trow = i0 + smt * 16 + r + 8 * half;
        int scol = j0 + jt * 16 + m;
        if (scol >= trow) sacc[r] = 0.0f;
      }
    }
    {
      __bf16* lds_s = lds_k;  // S[64][64] bf16, reuse K slab
#pragma unroll
      for (int r = 0; r < 8; ++r)
        lds_s[(smt * 16 + r + 8 * half) * 64 + jt * 16 + m] = (__bf16)sacc[r];
    }
    __syncthreads();

    // ---- acc += S . V_tile (two K=32 steps over the 64 s-columns) ----
    const __bf16* lds_s = lds_k;
    v16bf a0 = load_fragA(lds_s + (mtile * 16 + m) * 64 + 0, half);
    v16bf a1 = load_fragA(lds_s + (mtile * 16 + m) * 64 + 32, half);
#pragma unroll
    for (int nt = 0; nt < 8; ++nt) {
      int dcol = dquad * 128 + nt * 16 + m;
      acc[nt] = wmma_bf16(a0, load_fragB(lds_vt + dcol * 64 + 0, half), acc[nt]);
      acc[nt] = wmma_bf16(a1, load_fragB(lds_vt + dcol * 64 + 32, half), acc[nt]);
    }
  }

  int h = bh & (NH_ - 1);
  float g = 1.0f / (1.0f + expf(-mgate[h]));
  float ysc = 1.0f - g;
#pragma unroll
  for (int nt = 0; nt < 8; ++nt) {
#pragma unroll
    for (int r = 0; r < 8; ++r) {
      int trow = i0 + mtile * 16 + r + 8 * half;
      int dcol = dquad * 128 + nt * 16 + m;
      out[((size_t)bh * T_ + trow) * D_ + dcol] = ysc * acc[nt][r];
    }
  }
}

// ---------------------------------------------------------------------------
// Kernel B1: Qm_raw / Km_raw = x_raw @ W^T   (M=BT, N=256, K=512)
// grid (BT/32, 8, 2), 128 threads (4 waves -> 4 of the 16x16 tiles)
// ---------------------------------------------------------------------------
__global__ __launch_bounds__(128) void gemm_qk_kernel(const float* __restrict__ x,
                                                      const float* __restrict__ wq,
                                                      const float* __restrict__ wk,
                                                      float* __restrict__ qm,
                                                      float* __restrict__ km) {
  __shared__ __align__(16) __bf16 xs[32 * 256];   // 16KB
  __shared__ __align__(16) __bf16 wsh[32 * 256];  // 16KB
  int tid = threadIdx.x, lane = tid & 31, wave = tid >> 5;
  int m = lane & 15, half = lane >> 4;
  int r0 = blockIdx.x * 32;
  int n0 = blockIdx.y * 32;
  const float* W = blockIdx.z ? wk : wq;
  float* outp = blockIdx.z ? km : qm;
  int mt = wave & 1, nt = wave >> 1;
  v8f acc = (v8f){};
  for (int kc = 0; kc < 2; ++kc) {
    __syncthreads();
    load_tile<32, 256, 128>(xs, x + (size_t)r0 * D_ + kc * 256, D_, tid);
    load_tile<32, 256, 128>(wsh, W + (size_t)n0 * D_ + kc * 256, D_, tid);
    __syncthreads();
#pragma unroll
    for (int k0 = 0; k0 < 256; k0 += 32) {
      v16bf a  = load_fragA(xs + (mt * 16 + m) * 256 + k0, half);
      v16bf bm = load_fragB(wsh + (nt * 16 + m) * 256 + k0, half);
      acc = wmma_bf16(a, bm, acc);
    }
  }
#pragma unroll
  for (int r = 0; r < 8; ++r)
    outp[(size_t)(r0 + mt * 16 + r + 8 * half) * N_ + n0 + nt * 16 + m] = acc[r];
}

// one wave per row: x <- x / max(||x||, 1e-12), rows 0..BT-1 = Qm, BT.. = Km
__global__ __launch_bounds__(256) void l2norm_kernel(float* qm, float* km) {
  int wave = threadIdx.x >> 5, lane = threadIdx.x & 31;
  int row = blockIdx.x * 8 + wave;
  float* p = (row < BT_) ? (qm + (size_t)row * N_) : (km + (size_t)(row - BT_) * N_);
  float v[8], ss = 0.0f;
#pragma unroll
  for (int i = 0; i < 8; ++i) { v[i] = p[lane + i * 32]; ss += v[i] * v[i]; }
#pragma unroll
  for (int off = 16; off > 0; off >>= 1) ss += __shfl_xor(ss, off, 32);
  float sc = 1.0f / fmaxf(sqrtf(ss), 1e-12f);
#pragma unroll
  for (int i = 0; i < 8; ++i) p[lane + i * 32] = v[i] * sc;
}

// beta[bt][h] = sigmoid(x_raw[bt] . beta_w[h])
__global__ __launch_bounds__(256) void beta_kernel(const float* __restrict__ x,
                                                   const float* __restrict__ bw,
                                                   float* __restrict__ beta) {
  int idx = blockIdx.x * 256 + threadIdx.x;  // BT*NH
  int h = idx & (NH_ - 1), bt = idx >> 3;
  const float* xr = x + (size_t)bt * D_;
  const float* w  = bw + (size_t)h * D_;
  float s = 0.0f;
  for (int dd = 0; dd < D_; ++dd) s += xr[dd] * w[dd];
  beta[idx] = 1.0f / (1.0f + expf(-s));
}

// ---------------------------------------------------------------------------
// Kernel C: y_memory = LN(Qm @ M0), out += gate * y_memory
// grid (T/32, B*NH), 256 threads (8 waves: mtile=w&1, dquad=w>>1)
// ---------------------------------------------------------------------------
__global__ __launch_bounds__(256) void ymem_kernel(const float* __restrict__ qm,
                                                   const float* __restrict__ M0,
                                                   const float* __restrict__ mgate,
                                                   float* __restrict__ out) {
  __shared__ __align__(16) __bf16 lds_qm[32 * 256];   // 16KB
  __shared__ __align__(16) __bf16 lds_m0t[512 * 32];  // 32KB  M0^T chunk [d][n]
  __shared__ float lds_sum[32];
  __shared__ float lds_sq[32];
  int tid = threadIdx.x, lane = tid & 31, wave = tid >> 5;
  int m = lane & 15, half = lane >> 4;
  int bh = blockIdx.y, b = bh >> 3;
  int t0 = blockIdx.x * 32;
  int mtile = wave & 1, dquad = wave >> 1;
  v8f acc[8];
#pragma unroll
  for (int i = 0; i < 8; ++i) acc[i] = (v8f){};
  if (tid < 32) { lds_sum[tid] = 0.0f; lds_sq[tid] = 0.0f; }

  load_tile<32, 256, 256>(lds_qm, qm + ((size_t)b * T_ + t0) * N_, N_, tid);
  const float* M0bh = M0 + (size_t)bh * N_ * D_;
  for (int c = 0; c < 8; ++c) {
    __syncthreads();
    load_tile_T<32, 512, 256>(lds_m0t, M0bh + (size_t)c * 32 * D_, D_, tid);
    __syncthreads();
    v16bf a = load_fragA(lds_qm + (mtile * 16 + m) * N_ + c * 32, half);
#pragma unroll
    for (int nt = 0; nt < 8; ++nt) {
      int dcol = dquad * 128 + nt * 16 + m;
      v16bf bm = load_fragB(lds_m0t + dcol * 32, half);
      acc[nt] = wmma_bf16(a, bm, acc[nt]);
    }
  }
  // LayerNorm over D per row (64 lane-partials per row via DS float atomics)
#pragma unroll
  for (int r = 0; r < 8; ++r) {
    float s1 = 0.0f, s2 = 0.0f;
#pragma unroll
    for (int nt = 0; nt < 8; ++nt) { float v = acc[nt][r]; s1 += v; s2 += v * v; }
    int row = mtile * 16 + r + 8 * half;
    atomicAdd(&lds_sum[row], s1);
    atomicAdd(&lds_sq[row], s2);
  }
  __syncthreads();
  int h = bh & (NH_ - 1);
  float g = 1.0f / (1.0f + expf(-mgate[h]));
#pragma unroll
  for (int r = 0; r < 8; ++r) {
    int row = mtile * 16 + r + 8 * half;
    float mean = lds_sum[row] * (1.0f / 512.0f);
    float var  = lds_sq[row] * (1.0f / 512.0f) - mean * mean;
    float rstd = rsqrtf(var + 1e-5f);
#pragma unroll
    for (int nt = 0; nt < 8; ++nt) {
      int dcol = dquad * 128 + nt * 16 + m;
      size_t oi = ((size_t)bh * T_ + t0 + row) * D_ + dcol;
      out[oi] += g * (acc[nt][r] - mean) * rstd;
    }
  }
}

// ---------------------------------------------------------------------------
// Kernel D: delta-rule scan -> M_new. Columns of M are independent:
// grid (D/32, B*NH), 256 threads; thread (wave w, lane l) owns
// M[w*32 .. w*32+31][d = blk*32+l] in 32 VGPRs; sequential only over T.
// k-vector staging uses the CDNA5 async global->LDS path when available.
// ---------------------------------------------------------------------------
__global__ __launch_bounds__(256) void scan_kernel(const float* __restrict__ km,
                                                   const float* __restrict__ beta,
                                                   const float* __restrict__ xnext,
                                                   const float* __restrict__ M0,
                                                   float* __restrict__ Mout) {
  __shared__ float lds_kv[256];
  __shared__ float lds_part[256];
  __shared__ float lds_delta[32];
  int tid = threadIdx.x, lane = tid & 31, wave = tid >> 5;
  int bh = blockIdx.y, b = bh >> 3, h = bh & (NH_ - 1);
  int d = blockIdx.x * 32 + lane;
  const float* M0p = M0 + (size_t)bh * N_ * D_;
  float M[32];
#pragma unroll
  for (int i = 0; i < 32; ++i) M[i] = M0p[(size_t)(wave * 32 + i) * D_ + d];

  for (int t = 0; t < T_; ++t) {
    __syncthreads();
#if defined(HAVE_ASYNC_LDS)
    {
      typedef __attribute__((address_space(1))) int* as1ip;  // "__device__ int *"
      typedef __attribute__((address_space(3))) int* as3ip;  // "__shared__ int *"
      __builtin_amdgcn_global_load_async_to_lds_b32(
          (as1ip)(km + ((size_t)b * T_ + t) * N_ + tid), (as3ip)(&lds_kv[tid]), 0, 0);
#if __has_builtin(__builtin_amdgcn_s_wait_asynccnt)
      __builtin_amdgcn_s_wait_asynccnt(0);
#else
      asm volatile("s_wait_asynccnt 0x0" ::: "memory");
#endif
    }
#else
    lds_kv[tid] = km[((size_t)b * T_ + t) * N_ + tid];
#endif
    __syncthreads();
    float part = 0.0f;
#pragma unroll
    for (int i = 0; i < 32; ++i) part += lds_kv[wave * 32 + i] * M[i];
    lds_part[tid] = part;
    __syncthreads();
    if (wave == 0) {
      float vold = 0.0f;
#pragma unroll
      for (int w2 = 0; w2 < 8; ++w2) vold += lds_part[w2 * 32 + lane];
      float bta = beta[((size_t)b * T_ + t) * NH_ + h];
      float vv  = xnext[((size_t)b * T_ + t) * D_ + d];
      lds_delta[lane] = bta * (vv - vold);  // v_new - v_old
    }
    __syncthreads();
    float delta = lds_delta[lane];
#pragma unroll
    for (int i = 0; i < 32; ++i) M[i] += lds_kv[wave * 32 + i] * delta;
  }
  float* op = Mout + (size_t)bh * N_ * D_;
#pragma unroll
  for (int i = 0; i < 32; ++i) op[(size_t)(wave * 32 + i) * D_ + d] = M[i];
}

// ---------------------------------------------------------------------------
extern "C" void kernel_launch(void* const* d_in, const int* in_sizes, int n_in,
                              void* d_out, int out_size, void* d_ws, size_t ws_size,
                              hipStream_t stream) {
  const float* Q      = (const float*)d_in[0];
  const float* V      = (const float*)d_in[1];
  const float* x_raw  = (const float*)d_in[2];
  const float* x_next = (const float*)d_in[3];
  const float* thQ    = (const float*)d_in[4];
  const float* thK    = (const float*)d_in[5];
  const float* bw     = (const float*)d_in[6];
  const float* mg     = (const float*)d_in[7];
  const float* M0     = (const float*)d_in[8];

  float* y    = (float*)d_out;                       // [B,NH,T,D]
  float* Mout = y + (size_t)B_ * NH_ * T_ * D_;      // [B,NH,N,D]

  float* qm  = (float*)d_ws;                         // [BT, N]
  float* km  = qm + (size_t)BT_ * N_;                // [BT, N]
  float* bta = km + (size_t)BT_ * N_;                // [BT, NH]

  gemm_qk_kernel<<<dim3(BT_ / 32, N_ / 32, 2), 128, 0, stream>>>(x_raw, thQ, thK, qm, km);
  l2norm_kernel<<<dim3(2 * BT_ / 8), 256, 0, stream>>>(qm, km);
  beta_kernel<<<dim3(BT_ * NH_ / 256), 256, 0, stream>>>(x_raw, bw, bta);
  attn_kernel<<<dim3(T_ / 64, B_ * NH_), 512, 0, stream>>>(Q, V, mg, y);
  ymem_kernel<<<dim3(T_ / 32, B_ * NH_), 256, 0, stream>>>(qm, M0, mg, y);
  scan_kernel<<<dim3(D_ / 32, B_ * NH_), 256, 0, stream>>>(km, bta, x_next, M0, Mout);
  (void)in_sizes; (void)n_in; (void)out_size; (void)ws_size;
}
